// GCN_15573551415443
// MI455X (gfx1250) — matrix operand order repaired and checked by reading, not compiled, not run on hardware
//
#include <hip/hip_runtime.h>

// ---------------------------------------------------------------------------
// GCN forward for MI455X (gfx1250, wave32, WMMA + Tensor Data Mover).
// Pipeline:
//   cvt  : x->xb(bf16), adj->adjb(bf16), W1->W1t(bf16,T), W2->W2t(bf16,T)
//   G1   : s1t = (xb @ W1)^T            M=65536 N=1024 K=512   (tstore)
//   G2   : h1  = relu(adjb @ s1 + b1)   M=1024  N=1024 K=1024  x64 batch
//   G3   : s2t = (h1 @ W2)^T            M=65536 N=512  K=1024  (tstore)
//   G4   : h2t = relu(adjb @ s2 + b2)^T M=1024  N=512  K=1024  x64 (tstore)
//   pool : out[b] = (sum_h Wl[h] * sum_{n<len} h2t[b][h][n]) / len + bl
// GEMMs: bf16 inputs, f32 WMMA accumulation (v_wmma_f32_16x16x32_bf16).
// Tile staging: TDM tensor_load_to_lds (double-buffered, TENSORcnt-tracked),
// with hardware LDS padding reproducing the 80B anti-conflict row stride.
// Fallback (no TDM builtin): round-1 global_load+ds_store staging.
// ---------------------------------------------------------------------------

typedef __attribute__((ext_vector_type(16))) __bf16 v16bf;
typedef __attribute__((ext_vector_type(8)))  float  v8f;
typedef __attribute__((ext_vector_type(4)))  unsigned int v4u;
typedef __attribute__((ext_vector_type(8)))  int v8i;
typedef __attribute__((ext_vector_type(4)))  int v4i;

#if __has_builtin(__builtin_amdgcn_tensor_load_to_lds) && \
    __has_builtin(__builtin_amdgcn_s_wait_tensorcnt)
#define USE_TDM 1
#else
#define USE_TDM 0
#endif

__device__ __forceinline__ unsigned short f32_to_bf16(float f) {
  unsigned int u = __float_as_uint(f);
  unsigned int r = (u + 0x7FFFu + ((u >> 16) & 1u)) >> 16;  // RNE
  return (unsigned short)r;
}
__device__ __forceinline__ float bf16_to_f32(unsigned short h) {
  return __uint_as_float(((unsigned int)h) << 16);
}

// ---------------- elementwise f32 -> bf16 (4 per thread) -------------------
__global__ void cvt_f32_bf16_vec4(const float* __restrict__ src,
                                  unsigned short* __restrict__ dst, long n4) {
  long i = (long)blockIdx.x * blockDim.x + threadIdx.x;
  if (i >= n4) return;
  float4 f = ((const float4*)src)[i];
  ushort2 lo; lo.x = f32_to_bf16(f.x); lo.y = f32_to_bf16(f.y);
  ushort2 hi; hi.x = f32_to_bf16(f.z); hi.y = f32_to_bf16(f.w);
  ((ushort2*)dst)[2 * i]     = lo;
  ((ushort2*)dst)[2 * i + 1] = hi;
}

// ------------- transpose + convert (weights, small) ------------------------
__global__ void cvt_transpose_bf16(const float* __restrict__ src,
                                   unsigned short* __restrict__ dst,
                                   int R, int C) {
  long i = (long)blockIdx.x * blockDim.x + threadIdx.x;
  if (i >= (long)R * C) return;
  int r = (int)(i / C), c = (int)(i % C);
  dst[(size_t)c * R + r] = f32_to_bf16(src[i]);
}

#define LSTR 40  // LDS row stride in halves (80B: 16B-aligned, conflict-free)

#if USE_TDM
// Issue one TDM 2-D tile load: 128 rows x 32 bf16, row pitch = ld elements,
// into LDS at lds_off with pad 16B after every 64B row => 80B LDS row stride.
// D# fields per CDNA5 ISA ch.8 (group0 128b, group1 256b; groups 2/3 unused).
// Toolchain here is the 6-arg builtin: (v4u, v8i, v4i, v4i, v8i, i32 cpol).
__device__ __forceinline__ void tdm_load_tile_128x32(
    unsigned lds_off, const unsigned short* gptr, long ld_elems) {
  unsigned long long ga = (unsigned long long)(size_t)gptr;
  v4u g0;
  g0.x = 1u;                                   // count=1, user mode
  g0.y = lds_off;                              // lds_addr (bytes)
  g0.z = (unsigned)ga;                         // global_addr[31:0]
  g0.w = (unsigned)(ga >> 32) | 0x80000000u;   // global_addr[56:32] | type=2
  v8i g1;
  // data_size=1(2B) | pad_enable | pad_interval=3(16DW=64B) | pad_amount=3(4DW)
  g1[0] = (int)((1u << 16) | (1u << 20) | (3u << 22) | (3u << 25));
  g1[1] = (int)(32u << 16);                    // tensor_dim0 = 32 (tile-exact)
  g1[2] = (int)(128u << 16);                   // tensor_dim1 = 128
  g1[3] = (int)(32u << 16);                    // tile_dim0 = 32
  g1[4] = 128;                                 // tile_dim1 = 128 (tile_dim2=0)
  g1[5] = (int)(unsigned)ld_elems;             // tensor_dim0_stride[31:0]
  g1[6] = 0;                                   // stride hi / dim1_stride lo
  g1[7] = 0;
  v4i z4 = {0, 0, 0, 0};                       // groups 2/3: <=2D tensor
  v8i z8 = {0, 0, 0, 0, 0, 0, 0, 0};
  __builtin_amdgcn_tensor_load_to_lds(g0, g1, z4, z4, z8, 0);
}
#endif

// ---------------------------------------------------------------------------
// Batched TN GEMM: C = act(A @ B + bias)
//   A : [M][K] bf16 row-major         (lda, batch stride sA)
//   BT: [N][K] bf16 row-major (B^T)   (ldb, batch stride sB)
//   C : tstore ? Ct[n*ldc+m] : C[m*ldc+n], bf16 (sC batch stride)
// Block tile 128x128, K-step 32. 256 threads = 8 wave32 waves laid out 2x4;
// each wave owns a 64x32 sub-tile = 4x2 grid of 16x16 WMMA accumulators.
// M,N multiples of 128; K multiple of 32 (true for all calls here).
// ---------------------------------------------------------------------------
__global__ __launch_bounds__(256) void gemm_bf16_wmma(
    const unsigned short* __restrict__ A,
    const unsigned short* __restrict__ BT,
    unsigned short* __restrict__ C,
    int K, long lda, long ldb, long ldc,
    long sA, long sB, long sC,
    const float* __restrict__ bias, int relu, int tstore) {
  __shared__ __align__(16) unsigned short As[2][128 * LSTR];
  __shared__ __align__(16) unsigned short Bs[2][128 * LSTR];

  const int tid  = threadIdx.x;
  const int lane = tid & 31;
  const int wave = tid >> 5;
  const int wrow = wave >> 2;   // 0..1 -> 64-row slab
  const int wcol = wave & 3;    // 0..3 -> 32-col slab
  const int lr   = lane & 15;
  const int lhi  = lane >> 4;   // 0/1 (half-wave)

  const long m0 = (long)blockIdx.x * 128;
  const long n0 = (long)blockIdx.y * 128;
  const long z  = blockIdx.z;

  const unsigned short* Abase = A + z * sA + m0 * lda;   // tile row 0, col kt
  const unsigned short* Bbase = BT + z * sB + n0 * ldb;

  v8f acc[4][2] = {};

  // WMMA 16-bit operand layout (ISA 7.12.2):
  //  A frag lane: row m=lr; halves 0-7 = K[a0..a0+7], 8-15 = K[a0+16..a0+23],
  //               a0 = lhi*8.
  //  B frag lane: col n=lr; halves 0-7 = K[b0..b0+7], 8-15 = K[b0+16..b0+23],
  //               b0 = lhi*8.
  const int aoff = lhi * 8;
  const int boff = lhi * 8;
  const int ntiles = K >> 5;

#if USE_TDM
  // ---- TDM double-buffered staging: wave 0 drives the Tensor Data Mover ----
  if (tid < 32) {
    tdm_load_tile_128x32((unsigned)(size_t)&As[0][0], Abase, lda);
    tdm_load_tile_128x32((unsigned)(size_t)&Bs[0][0], Bbase, ldb);
  }
  for (int it = 0; it < ntiles; ++it) {
    const int cur = it & 1;
    if (tid < 32) {
      if (it + 1 < ntiles) {
        const long kn = (long)(it + 1) << 5;
        tdm_load_tile_128x32((unsigned)(size_t)&As[cur ^ 1][0], Abase + kn, lda);
        tdm_load_tile_128x32((unsigned)(size_t)&Bs[cur ^ 1][0], Bbase + kn, ldb);
        __builtin_amdgcn_s_wait_tensorcnt(2);  // oldest 2 (current tile) done
      } else {
        __builtin_amdgcn_s_wait_tensorcnt(0);
      }
    }
    __syncthreads();  // current tile visible to all waves

    const unsigned short* Asb = &As[cur][0];
    const unsigned short* Bsb = &Bs[cur][0];
    union { uint4 q[2]; v16bf v; } af[4], bfr[2];
#pragma unroll
    for (int i = 0; i < 4; ++i) {
      const int r = wrow * 64 + i * 16 + lr;
      af[i].q[0] = *(const uint4*)&Asb[r * LSTR + aoff];
      af[i].q[1] = *(const uint4*)&Asb[r * LSTR + aoff + 16];
    }
#pragma unroll
    for (int j = 0; j < 2; ++j) {
      const int c = wcol * 32 + j * 16 + lr;
      bfr[j].q[0] = *(const uint4*)&Bsb[c * LSTR + boff];
      bfr[j].q[1] = *(const uint4*)&Bsb[c * LSTR + boff + 16];
    }
#pragma unroll
    for (int i = 0; i < 4; ++i)
#pragma unroll
      for (int j = 0; j < 2; ++j)
        acc[i][j] = __builtin_amdgcn_wmma_f32_16x16x32_bf16(
            false, af[i].v, false, bfr[j].v, (short)0, acc[i][j], false, false);

    __syncthreads();  // reads done before this buffer is DMA'd again
  }
#else
  // ---- fallback: cooperative global_load + ds_store staging (round 1) -----
  const unsigned short* Ag = Abase + (long)(tid >> 1) * lda + (long)((tid & 1) * 16);
  const unsigned short* Bg = Bbase + (long)(tid >> 1) * ldb + (long)((tid & 1) * 16);
  const int ldsOff = (tid >> 1) * LSTR + (tid & 1) * 16;

  for (int it = 0; it < ntiles; ++it) {
    const int kt = it << 5;
    uint4 a0 = *(const uint4*)(Ag + kt);
    uint4 a1 = *(const uint4*)(Ag + kt + 8);
    uint4 b0 = *(const uint4*)(Bg + kt);
    uint4 b1 = *(const uint4*)(Bg + kt + 8);
    __syncthreads();
    *(uint4*)&As[0][ldsOff]     = a0;
    *(uint4*)&As[0][ldsOff + 8] = a1;
    *(uint4*)&Bs[0][ldsOff]     = b0;
    *(uint4*)&Bs[0][ldsOff + 8] = b1;
    __syncthreads();

    union { uint4 q[2]; v16bf v; } af[4], bfr[2];
#pragma unroll
    for (int i = 0; i < 4; ++i) {
      const int r = wrow * 64 + i * 16 + lr;
      af[i].q[0] = *(const uint4*)&As[0][r * LSTR + aoff];
      af[i].q[1] = *(const uint4*)&As[0][r * LSTR + aoff + 16];
    }
#pragma unroll
    for (int j = 0; j < 2; ++j) {
      const int c = wcol * 32 + j * 16 + lr;
      bfr[j].q[0] = *(const uint4*)&Bs[0][c * LSTR + boff];
      bfr[j].q[1] = *(const uint4*)&Bs[0][c * LSTR + boff + 16];
    }
#pragma unroll
    for (int i = 0; i < 4; ++i)
#pragma unroll
      for (int j = 0; j < 2; ++j)
        acc[i][j] = __builtin_amdgcn_wmma_f32_16x16x32_bf16(
            false, af[i].v, false, bfr[j].v, (short)0, acc[i][j], false, false);
  }
#endif

  // Epilogue. C/D layout: lane holds column n=lr; rows v + 8*lhi per tile.
#pragma unroll
  for (int j = 0; j < 2; ++j) {
    const long n = n0 + wcol * 32 + j * 16 + lr;
    const float bv = bias ? bias[n] : 0.0f;
#pragma unroll
    for (int i = 0; i < 4; ++i) {
      const long mbase = m0 + wrow * 64 + i * 16 + 8 * lhi;
      float v[8];
#pragma unroll
      for (int t = 0; t < 8; ++t) {
        float x = acc[i][j][t] + bv;
        v[t] = relu ? fmaxf(x, 0.0f) : x;
      }
      if (tstore) {
        // transposed store: 8 consecutive m for fixed n -> one 16B store
        union { uint4 q; unsigned short s[8]; } pk;
#pragma unroll
        for (int t = 0; t < 8; ++t) pk.s[t] = f32_to_bf16(v[t]);
        *(uint4*)&C[z * sC + n * ldc + mbase] = pk.q;
      } else {
#pragma unroll
        for (int t = 0; t < 8; ++t)
          C[z * sC + (mbase + t) * ldc + n] = f32_to_bf16(v[t]);
      }
    }
  }
}

// ------------- masked mean-pool + linear head ------------------------------
// h2t: [B][H][Nn] bf16 (feature-major => contiguous row reduction per h)
__global__ void pool_head(const unsigned short* __restrict__ h2t,
                          const int* __restrict__ length,
                          const float* __restrict__ Wl,
                          const float* __restrict__ bl,
                          float* __restrict__ out, int H, int Nn) {
  const int b = blockIdx.x;
  const int len = length[b];
  const unsigned short* base = h2t + (size_t)b * H * Nn;
  float acc = 0.0f;
  for (int h = threadIdx.x; h < H; h += blockDim.x) {
    const unsigned short* row = base + (size_t)h * Nn;
    float s = 0.0f;
    for (int n = 0; n < len; ++n) s += bf16_to_f32(row[n]);
    acc += s * Wl[h];
  }
  __shared__ float red[256];
  red[threadIdx.x] = acc;
  __syncthreads();
  for (int off = 128; off > 0; off >>= 1) {
    if ((int)threadIdx.x < off) red[threadIdx.x] += red[threadIdx.x + off];
    __syncthreads();
  }
  if (threadIdx.x == 0) out[b] = red[0] / (float)len + bl[0];
}

// ---------------------------------------------------------------------------
extern "C" void kernel_launch(void* const* d_in, const int* in_sizes, int n_in,
                              void* d_out, int out_size, void* d_ws,
                              size_t ws_size, hipStream_t stream) {
  const float* x      = (const float*)d_in[0];  // [64,1024,512]
  const float* adj    = (const float*)d_in[1];  // [64,1024,1024]
  const int*   length = (const int*)  d_in[2];  // [64]
  const float* W1     = (const float*)d_in[3];  // [512,1024]
  const float* b1     = (const float*)d_in[4];  // [1024]
  const float* W2     = (const float*)d_in[5];  // [1024,512]
  const float* b2     = (const float*)d_in[6];  // [512]
  const float* Wl     = (const float*)d_in[7];  // [512,1]
  const float* bl     = (const float*)d_in[8];  // [1]
  float* out = (float*)d_out;                   // [64]

  // Workspace layout (bytes), 450 MB peak with aliasing.
  char* ws = (char*)d_ws;
  unsigned short* adjb = (unsigned short*)(ws + 0);          // 134,217,728 B
  unsigned short* xb   = (unsigned short*)(ws + 134217728);  //  67,108,864 B
  unsigned short* s1t  = (unsigned short*)(ws + 201326592);  // 134,217,728 B
  unsigned short* h1   = (unsigned short*)(ws + 335544320);  // 134,217,728 B
  unsigned short* w1t  = (unsigned short*)(ws + 469762048);  //   1,048,576 B
  unsigned short* w2t  = (unsigned short*)(ws + 470810624);  //   1,048,576 B
  unsigned short* s2t  = xb;   // alias: xb dead after G1
  unsigned short* h2t  = s1t;  // alias: s1t dead after G2

  // -------- conversions --------
  {
    long n4 = (64L * 1024 * 512) / 4;  // 8,388,608
    cvt_f32_bf16_vec4<<<(unsigned)(n4 / 256), 256, 0, stream>>>(x, xb, n4);
  }
  {
    long n4 = (64L * 1024 * 1024) / 4;  // 16,777,216
    cvt_f32_bf16_vec4<<<(unsigned)(n4 / 256), 256, 0, stream>>>(adj, adjb, n4);
  }
  cvt_transpose_bf16<<<(512 * 1024 + 255) / 256, 256, 0, stream>>>(W1, w1t, 512, 1024);
  cvt_transpose_bf16<<<(1024 * 512 + 255) / 256, 256, 0, stream>>>(W2, w2t, 1024, 512);

  dim3 blk(256);

  // G1: s1t[h][b*1024+node] = (x @ W1)^T   M=65536 N=1024 K=512
  gemm_bf16_wmma<<<dim3(65536 / 128, 1024 / 128, 1), blk, 0, stream>>>(
      xb, w1t, s1t, 512, 512, 512, 65536, 0, 0, 0, nullptr, 0, 1);

  // G2: h1[b][node][h] = relu(adj @ s1 + b1)   M=1024 N=1024 K=1024, 64 batches
  gemm_bf16_wmma<<<dim3(1024 / 128, 1024 / 128, 64), blk, 0, stream>>>(
      adjb, s1t, h1, 1024, 1024, 65536, 1024, 1048576, 1024, 1048576, b1, 1, 0);

  // G3: s2t[h2][b*1024+node] = (h1 @ W2)^T   M=65536 N=512 K=1024
  gemm_bf16_wmma<<<dim3(65536 / 128, 512 / 128, 1), blk, 0, stream>>>(
      h1, w2t, s2t, 1024, 1024, 1024, 65536, 0, 0, 0, nullptr, 0, 1);

  // G4: h2t[b][h2][node] = relu(adj @ s2 + b2)^T  M=1024 N=512 K=1024, 64 b.
  gemm_bf16_wmma<<<dim3(1024 / 128, 512 / 128, 64), blk, 0, stream>>>(
      adjb, s2t, h2t, 1024, 1024, 65536, 1024, 1048576, 1024, 524288, b2, 1, 1);

  // pool + head
  pool_head<<<64, 256, 0, stream>>>(h2t, length, Wl, bl, out, 512, 1024);
}
